// GloAttnConv_90649579749715
// MI455X (gfx1250) — matrix operand order repaired and checked by reading, not compiled, not run on hardware
//
#include <hip/hip_runtime.h>
#include <stdint.h>
#include <stddef.h>

// ---------------------------------------------------------------------------
// GloAttnConv for MI455X (gfx1250): WMMA bf16 GEMMs + TDM async staging +
// atomic edge scatter. Wave32; WMMA A/B fragments use contiguous k = j + 16*(lane/16).
// ---------------------------------------------------------------------------

#define DDIM   64
#define HDIM   128     // NUM_HEADS * D = 2*64
#define NHEADS 2
#define KORDER 4
#define MAXNODE_TILES 125   // ceil(2000/16), reference's static max_node

typedef __attribute__((ext_vector_type(16))) __bf16 v16bf;
typedef __attribute__((ext_vector_type(8)))  float  v8f;
typedef unsigned int u32x4 __attribute__((ext_vector_type(4)));
typedef int          i32x4 __attribute__((ext_vector_type(4)));
typedef int          i32x8 __attribute__((ext_vector_type(8)));

#define CDIV(a, b) (((a) + (b) - 1) / (b))

// Tensor Data Mover availability (arity differs across toolchains)
#if defined(__has_builtin)
#if __has_builtin(__builtin_amdgcn_tensor_load_to_lds) && \
    __has_builtin(__builtin_amdgcn_s_wait_tensorcnt)
#define HAVE_TDM 1
#endif
#endif
#ifndef HAVE_TDM
#define HAVE_TDM 0
#endif

// ---- bf16 helpers (storage = ushort, RNE convert) -------------------------
__device__ __forceinline__ unsigned short f2bf(float f) {
  unsigned int x = __float_as_uint(f);
  x += 0x7FFFu + ((x >> 16) & 1u);
  return (unsigned short)(x >> 16);
}
__device__ __forceinline__ float bf2f(unsigned short u) {
  return __uint_as_float(((unsigned int)u) << 16);
}
__device__ __forceinline__ __bf16 bfbits(unsigned short u) {
  union { unsigned short u; __bf16 b; } c; c.u = u; return c.b;
}

#if HAVE_TDM
// ---- TDM: async-copy a 2D bf16 tile [rows<=32, 64 cols] global->LDS -------
// Tensor descriptor per CDNA5 ISA.
// tensor_dim0=64 elems, tensor_dim1=rem (rows beyond rem zero-fill the LDS),
// tile = 64 x 32, row stride = HDIM elems, data_size=2B.
__device__ __forceinline__ void tdm_load_tile(unsigned lds_addr, const void* gaddr,
                                              unsigned rem_rows) {
  unsigned long long ga = (unsigned long long)gaddr;
  u32x4 g0;
  g0[0] = 1u;                                              // count=1, user mode
  g0[1] = lds_addr;                                        // LDS byte address
  g0[2] = (unsigned)(ga & 0xFFFFFFFFu);                    // global_addr[31:0]
  g0[3] = (unsigned)((ga >> 32) & 0x01FFFFFFu) | (2u << 30); // ga[56:32] | type=2
  i32x8 g1;
  g1[0] = (int)(1u << 16);                 // wg_mask=0, data_size=1 (2 bytes)
  g1[1] = (int)(64u << 16);                // tensor_dim0[15:0]=64 (bits 63:48)
  g1[2] = (int)((rem_rows & 0xFFFFu) << 16); // dim0 hi=0 | tensor_dim1 lo
  g1[3] = (int)((rem_rows >> 16) | (64u << 16)); // tensor_dim1 hi | tile_dim0=64
  g1[4] = 32;                              // tile_dim1=32, tile_dim2=0
  g1[5] = (int)HDIM;                       // tensor_dim0_stride (elems), low 32
  g1[6] = 0;                               // stride0 hi | stride1 lo
  g1[7] = 0;                               // stride1 hi
  i32x4 z4 = {0, 0, 0, 0};
#if __clang_major__ >= 23
  i32x8 z8 = {0, 0, 0, 0, 0, 0, 0, 0};
  __builtin_amdgcn_tensor_load_to_lds(g0, g1, z4, z4, z8, 0);
#else
  __builtin_amdgcn_tensor_load_to_lds(g0, g1, z4, z4, 0);
#endif
}
#endif

// ---- tiny utility kernels -------------------------------------------------
__global__ void k_prefix(const int* __restrict__ nn, int B, int* __restrict__ ptr) {
  if (blockIdx.x == 0 && threadIdx.x == 0) {
    int s = 0; ptr[0] = 0;
    for (int b = 0; b < B; ++b) { s += nn[b]; ptr[b + 1] = s; }
  }
}

__global__ void k_batch(const int* __restrict__ ptr, int B, int N, int* __restrict__ batch) {
  int i = blockIdx.x * blockDim.x + threadIdx.x;
  if (i >= N) return;
  int lo = 0, hi = B;
  while (hi - lo > 1) { int mid = (lo + hi) >> 1; if (ptr[mid] <= i) lo = mid; else hi = mid; }
  batch[i] = lo;
}

__global__ void k_f32_to_bf16(const float* __restrict__ src, unsigned short* __restrict__ dst, int n) {
  int i = blockIdx.x * blockDim.x + threadIdx.x;
  if (i < n) dst[i] = f2bf(src[i]);
}

__global__ void k_zero_f32(float* __restrict__ p, int n) {
  int i = blockIdx.x * blockDim.x + threadIdx.x;
  if (i < n) p[i] = 0.0f;
}

// ---- generic WMMA GEMM: out[n,c] = (sum_k A[n,k]*W[c,k] + bias[c])*scale --
__global__ void k_gemm_bias(const unsigned short* __restrict__ A,
                            const unsigned short* __restrict__ W,
                            const float* __restrict__ bias,
                            float* __restrict__ out,
                            int Nrows, int Kd, int Cols, float scale) {
  int warp = threadIdx.x >> 5;
  int tile = blockIdx.x * (blockDim.x >> 5) + warp;
  int colTiles = Cols >> 4;
  int rowTile = tile / colTiles;
  int colTile = tile - rowTile * colTiles;
  if (rowTile >= CDIV(Nrows, 16)) return;
  int lane = threadIdx.x & 31, lmod = lane & 15, lhalf = lane >> 4;
  int rowBase = rowTile << 4, colBase = colTile << 4;

  v8f c = {};
  for (int k0 = 0; k0 < Kd; k0 += 32) {
    v16bf a, b;
    int rrow = rowBase + lmod; if (rrow >= Nrows) rrow = Nrows - 1;
#pragma unroll
    for (int j = 0; j < 16; ++j) {
      int ka = j + 16 * lhalf;                              // A: M=lane%16, K=j+16*(lane/16)
      a[j] = bfbits(A[(size_t)rrow * Kd + k0 + ka]);
      int kb = j + 16 * lhalf;                              // B: N=lane%16, K=j+16*(lane/16)
      b[j] = bfbits(W[(size_t)(colBase + lmod) * Kd + k0 + kb]);
    }
    c = __builtin_amdgcn_wmma_f32_16x16x32_bf16(false, a, false, b, (short)0, c, false, false);
  }
  float bv = bias[colBase + lmod];
#pragma unroll
  for (int r = 0; r < 8; ++r) {
    int m = rowBase + r + 8 * lhalf;                        // C/D: M=r+8*(lane/16)
    if (m < Nrows) out[(size_t)m * Cols + colBase + lmod] = (c[r] + bv) * scale;
  }
}

// ---- per-(n,h) L2 normalization, write bf16 (wave32 butterfly) ------------
__global__ void k_normalize(const float* __restrict__ q, const float* __restrict__ k,
                            unsigned short* __restrict__ qs, unsigned short* __restrict__ kk,
                            int nRows /* = N*NHEADS */) {
  int warp = (blockIdx.x * blockDim.x + threadIdx.x) >> 5;
  int lane = threadIdx.x & 31;
  if (warp >= nRows) return;
  size_t base = (size_t)warp * DDIM;  // [N,128] == [(n*2+h), 64]
  float v0 = q[base + lane], v1 = q[base + lane + 32];
  float ss = v0 * v0 + v1 * v1;
  for (int m = 16; m; m >>= 1) ss += __shfl_xor(ss, m, 32);
  float inv = rsqrtf(ss);
  qs[base + lane] = f2bf(v0 * inv); qs[base + lane + 32] = f2bf(v1 * inv);
  v0 = k[base + lane]; v1 = k[base + lane + 32];
  ss = v0 * v0 + v1 * v1;
  for (int m = 16; m; m >>= 1) ss += __shfl_xor(ss, m, 32);
  inv = rsqrtf(ss);
  kk[base + lane] = f2bf(v0 * inv); kk[base + lane + 32] = f2bf(v1 * inv);
}

// ---- degree / edge weights / loop-invariant edge message ------------------
__global__ void k_count_deg(const int* __restrict__ row, int E, float* __restrict__ deg) {
  int e = blockIdx.x * blockDim.x + threadIdx.x;
  if (e < E) atomicAdd(&deg[row[e]], 1.0f);
}

__global__ void k_edge_w(const int* __restrict__ row, const int* __restrict__ col,
                         const float* __restrict__ deg, float* __restrict__ wE, int E) {
  int e = blockIdx.x * blockDim.x + threadIdx.x;
  if (e >= E) return;
  float dr = deg[row[e]], dc = deg[col[e]];
  float a = dr > 0.0f ? rsqrtf(dr) : 0.0f;
  float b = dc > 0.0f ? rsqrtf(dc) : 0.0f;
  wE[e] = a * b;
}

__global__ void k_msg(const int* __restrict__ row, const int* __restrict__ eattr,
                      const float* __restrict__ bond, const float* __restrict__ wE,
                      float* __restrict__ msg, int E) {
  long long idx = (long long)blockIdx.x * blockDim.x + threadIdx.x;
  if (idx >= (long long)E * DDIM) return;
  int e = (int)(idx >> 6), d = (int)(idx & 63);
  float w = wE[e];
  if (w == 0.0f) return;
  int a0 = eattr[e * 3 + 0], a1 = eattr[e * 3 + 1], a2 = eattr[e * 3 + 2];
  float v = bond[(0 * 8 + a0) * DDIM + d] + bond[(1 * 8 + a1) * DDIM + d] +
            bond[(2 * 8 + a2) * DDIM + d];
  atomicAdd(&msg[(size_t)row[e] * DDIM + d], w * v);
}

// ---- init cur = acc = broadcast(x) ----------------------------------------
__global__ void k_init(const float* __restrict__ x, float* __restrict__ cur,
                       float* __restrict__ acc, unsigned short* __restrict__ curbf, int N) {
  int i = blockIdx.x * blockDim.x + threadIdx.x;
  if (i >= N * HDIM) return;
  int n = i >> 7, d = i & 127;
  float v = x[(size_t)n * DDIM + (d & 63)];
  cur[i] = v; acc[i] = v; curbf[i] = f2bf(v);
}

// ---- GCN scatter: gcn[row] += w * cur[col] --------------------------------
__global__ void k_gcn(const int* __restrict__ row, const int* __restrict__ col,
                      const float* __restrict__ wE, const float* __restrict__ cur,
                      float* __restrict__ gcn, int E) {
  long long idx = (long long)blockIdx.x * blockDim.x + threadIdx.x;
  if (idx >= (long long)E * HDIM) return;
  int e = (int)(idx >> 7), d = (int)(idx & 127);
  float w = wE[e];
  if (w == 0.0f) return;
  atomicAdd(&gcn[(size_t)row[e] * HDIM + d], w * cur[(size_t)col[e] * HDIM + d]);
}

// ---- per-graph k_sum / v_sum ----------------------------------------------
__global__ void k_sums(const int* __restrict__ ptr, const unsigned short* __restrict__ kk,
                       const float* __restrict__ cur, float* __restrict__ ksum,
                       float* __restrict__ vsum, int B) {
  int i = blockIdx.x * blockDim.x + threadIdx.x;
  if (i >= B * HDIM) return;
  int b = i >> 7, d = i & 127;
  int s = ptr[b], e = ptr[b + 1];
  float ks = 0.0f, vs = 0.0f;
  for (int n = s; n < e; ++n) {
    ks += bf2f(kk[(size_t)n * HDIM + d]);
    vs += cur[(size_t)n * HDIM + d];
  }
  ksum[i] = ks; vsum[i] = vs;
}

// ---- per-(graph,head) KV = K^T V via WMMA, operands staged by TDM ---------
__global__ void k_kv(const int* __restrict__ ptr, const unsigned short* __restrict__ kk,
                     const unsigned short* __restrict__ curbf, float* __restrict__ kv, int B) {
  int bh = blockIdx.x;                 // B*NHEADS blocks
  int b = bh >> 1, h = bh & 1;
  int warp = threadIdx.x >> 5;         // 4 waves: row tiles of KV (d_k dim)
  int mT = warp;
  int lane = threadIdx.x & 31, lmod = lane & 15, lhalf = lane >> 4;
  int s = ptr[b], e = ptr[b + 1];
  int hb = h * DDIM;
  v8f cacc[4] = {};

#if HAVE_TDM
  // Stage one 32-node chunk of K and V head-slices in LDS via Tensor DMA.
  __shared__ unsigned short kkS[32 * DDIM];   // [node][d_k]  (4 KB)
  __shared__ unsigned short vS[32 * DDIM];    // [node][d_v]  (4 KB)
  unsigned lds_kk = (unsigned)(unsigned long long)(void*)&kkS[0];
  unsigned lds_v  = (unsigned)(unsigned long long)(void*)&vS[0];

  for (int n0 = s; n0 < e; n0 += 32) {
    unsigned rem = (unsigned)(e - n0);   // rows >= rem are zero-filled by TDM
    __syncthreads();                     // prev chunk fully consumed
    if (warp == 0) {
      tdm_load_tile(lds_kk, kk + (size_t)n0 * HDIM + hb, rem);
      tdm_load_tile(lds_v,  curbf + (size_t)n0 * HDIM + hb, rem);
      __builtin_amdgcn_s_wait_tensorcnt(0);
    }
    __syncthreads();                     // LDS tiles visible to all waves
    v16bf a;
#pragma unroll
    for (int j = 0; j < 16; ++j) {       // A[m=d_k][k=node] = K^T (from LDS)
      int nl = j + 16 * lhalf;
      a[j] = bfbits(kkS[nl * DDIM + mT * 16 + lmod]);
    }
#pragma unroll
    for (int cT = 0; cT < 4; ++cT) {
      v16bf bfr;
#pragma unroll
      for (int j = 0; j < 16; ++j) {     // B[k=node][c=e_dim] = V (from LDS)
        int nl = j + 16 * lhalf;
        bfr[j] = bfbits(vS[nl * DDIM + cT * 16 + lmod]);
      }
      cacc[cT] = __builtin_amdgcn_wmma_f32_16x16x32_bf16(false, a, false, bfr, (short)0,
                                                         cacc[cT], false, false);
    }
  }
#else
  for (int n0 = s; n0 < e; n0 += 32) {
    if (n0 + 32 < e)
      __builtin_prefetch(curbf + (size_t)(n0 + 32) * HDIM, 0, 0);
    v16bf a;
#pragma unroll
    for (int j = 0; j < 16; ++j) {
      int n = n0 + j + 16 * lhalf;
      a[j] = bfbits(n < e ? kk[(size_t)n * HDIM + hb + mT * 16 + lmod] : (unsigned short)0);
    }
#pragma unroll
    for (int cT = 0; cT < 4; ++cT) {
      v16bf bfr;
#pragma unroll
      for (int j = 0; j < 16; ++j) {
        int n = n0 + j + 16 * lhalf;
        bfr[j] = bfbits(n < e ? curbf[(size_t)n * HDIM + hb + cT * 16 + lmod] : (unsigned short)0);
      }
      cacc[cT] = __builtin_amdgcn_wmma_f32_16x16x32_bf16(false, a, false, bfr, (short)0,
                                                         cacc[cT], false, false);
    }
  }
#endif

  size_t kvbase = (size_t)bh * DDIM * DDIM;
#pragma unroll
  for (int cT = 0; cT < 4; ++cT)
#pragma unroll
    for (int r = 0; r < 8; ++r) {
      int m = mT * 16 + r + 8 * lhalf;
      kv[kvbase + (size_t)m * DDIM + cT * 16 + lmod] = cacc[cT][r];
    }
}

// ---- numer = Q @ KV via WMMA (one block per (graph, nodeTile)) ------------
__global__ void k_qkv(const int* __restrict__ ptr, const unsigned short* __restrict__ qs,
                      const unsigned short* __restrict__ kvbf, float* __restrict__ numer,
                      int maxTiles) {
  int b = blockIdx.x / maxTiles, t = blockIdx.x - b * maxTiles;
  int s = ptr[b], e = ptr[b + 1];
  int n0 = s + t * 16;
  if (n0 >= e) return;
  int warp = threadIdx.x >> 5, lane = threadIdx.x & 31, lmod = lane & 15, lhalf = lane >> 4;

#pragma unroll
  for (int pp = 0; pp < 2; ++pp) {     // 4 waves x 2 = 8 (head, eTile) pairs
    int p = warp * 2 + pp;
    int h = p >> 2, eT = p & 3;
    int hb = h * DDIM;
    v8f c = {};
    for (int k0 = 0; k0 < DDIM; k0 += 32) {
      v16bf a, bfr;
      int n = n0 + lmod;
#pragma unroll
      for (int j = 0; j < 16; ++j) {   // A[m=node][k=d] = Q
        int d = k0 + j + 16 * lhalf;
        a[j] = bfbits(n < e ? qs[(size_t)n * HDIM + hb + d] : (unsigned short)0);
      }
#pragma unroll
      for (int j = 0; j < 16; ++j) {   // B[k=d][c=e_dim] = KV
        int d = k0 + j + 16 * lhalf;
        bfr[j] = bfbits(kvbf[((size_t)(b * 2 + h) * DDIM + d) * DDIM + eT * 16 + lmod]);
      }
      c = __builtin_amdgcn_wmma_f32_16x16x32_bf16(false, a, false, bfr, (short)0, c, false, false);
    }
#pragma unroll
    for (int r = 0; r < 8; ++r) {
      int n = n0 + r + 8 * lhalf;
      if (n < e) numer[(size_t)n * HDIM + hb + eT * 16 + lmod] = c[r];
    }
  }
}

// ---- combine: cur = 0.5*(gcn+msg) + 0.5*(numer+vsum)/denom; acc += cur ----
__global__ void k_combine(const unsigned short* __restrict__ qs, const float* __restrict__ ksum,
                          const float* __restrict__ vsum, const float* __restrict__ numer,
                          const float* __restrict__ gcn, const float* __restrict__ msg,
                          const int* __restrict__ batch, const int* __restrict__ nn,
                          float* __restrict__ cur, unsigned short* __restrict__ curbf,
                          float* __restrict__ acc) {
  int n = blockIdx.x;
  int d = threadIdx.x;               // 128 threads = one node
  int b = batch[n];
  size_t i = (size_t)n * HDIM + d;
  float qv = bf2f(qs[i]);
  __shared__ float sd[HDIM];
  sd[d] = qv * ksum[b * HDIM + d];
  __syncthreads();
  int h = d >> 6, dh = d & 63;
  for (int s = 32; s > 0; s >>= 1) { // per-head 64-wide tree reduce
    if (dh < s) sd[d] += sd[d + s];
    __syncthreads();
  }
  float denom = sd[h * 64] + (float)nn[b];
  float attn = (numer[i] + vsum[b * HDIM + d]) / denom;
  float g = gcn[i] + msg[(size_t)n * DDIM + dh];
  float cv = 0.5f * g + 0.5f * attn;
  cur[i] = cv; curbf[i] = f2bf(cv); acc[i] += cv;
}

// ---------------------------------------------------------------------------
extern "C" void kernel_launch(void* const* d_in, const int* in_sizes, int n_in,
                              void* d_out, int out_size, void* d_ws, size_t ws_size,
                              hipStream_t stream) {
  const float* x    = (const float*)d_in[0];
  const float* Wq_w = (const float*)d_in[1];
  const float* Wq_b = (const float*)d_in[2];
  const float* Wk_w = (const float*)d_in[3];
  const float* Wk_b = (const float*)d_in[4];
  const float* Wo_w = (const float*)d_in[5];
  const float* Wo_b = (const float*)d_in[6];
  const float* bond = (const float*)d_in[7];
  const int*   eidx = (const int*)d_in[8];
  const int*   eatt = (const int*)d_in[9];
  const int*   nn   = (const int*)d_in[10];

  const int N = in_sizes[0] / DDIM;
  const int E = in_sizes[8] / 2;
  const int B = in_sizes[10];
  const int* erow = eidx;
  const int* ecol = eidx + E;

  // ---- carve workspace ----
  char* ws = (char*)d_ws;
  size_t off = 0;
  auto carve = [&](size_t bytes) -> void* {
    void* p = ws + off;
    off += (bytes + 255) & ~(size_t)255;
    return p;
  };
  int*            ptrb   = (int*)carve((size_t)(B + 1) * 4);
  int*            batch  = (int*)carve((size_t)N * 4);
  unsigned short* x_bf   = (unsigned short*)carve((size_t)N * DDIM * 2);
  unsigned short* Wq_bf  = (unsigned short*)carve((size_t)HDIM * DDIM * 2);
  unsigned short* Wk_bf  = (unsigned short*)carve((size_t)HDIM * DDIM * 2);
  unsigned short* Wo_bf  = (unsigned short*)carve((size_t)DDIM * HDIM * 2);
  float*          q_f    = (float*)carve((size_t)N * HDIM * 4);   // reused as gcn
  float*          k_f    = (float*)carve((size_t)N * HDIM * 4);   // reused as numer
  unsigned short* qs_bf  = (unsigned short*)carve((size_t)N * HDIM * 2);
  unsigned short* kk_bf  = (unsigned short*)carve((size_t)N * HDIM * 2);
  float*          deg    = (float*)carve((size_t)N * 4);
  float*          wE     = (float*)carve((size_t)E * 4);
  float*          msg    = (float*)carve((size_t)N * DDIM * 4);
  float*          cur    = (float*)carve((size_t)N * HDIM * 4);
  unsigned short* cur_bf = (unsigned short*)carve((size_t)N * HDIM * 2); // reused for acc_bf
  float*          acc    = (float*)carve((size_t)N * HDIM * 4);
  float*          kv     = (float*)carve((size_t)B * NHEADS * DDIM * DDIM * 4);
  unsigned short* kv_bf  = (unsigned short*)carve((size_t)B * NHEADS * DDIM * DDIM * 2);
  float*          ksum   = (float*)carve((size_t)B * HDIM * 4);
  float*          vsum   = (float*)carve((size_t)B * HDIM * 4);
  (void)ws_size; (void)n_in; (void)out_size;

  const int tb = 256;

  // ---- loop-invariant prep ----
  k_prefix<<<1, 32, 0, stream>>>(nn, B, ptrb);
  k_batch<<<CDIV(N, tb), tb, 0, stream>>>(ptrb, B, N, batch);
  k_f32_to_bf16<<<CDIV(N * DDIM, tb), tb, 0, stream>>>(x, x_bf, N * DDIM);
  k_f32_to_bf16<<<CDIV(HDIM * DDIM, tb), tb, 0, stream>>>(Wq_w, Wq_bf, HDIM * DDIM);
  k_f32_to_bf16<<<CDIV(HDIM * DDIM, tb), tb, 0, stream>>>(Wk_w, Wk_bf, HDIM * DDIM);
  k_f32_to_bf16<<<CDIV(DDIM * HDIM, tb), tb, 0, stream>>>(Wo_w, Wo_bf, DDIM * HDIM);

  int projTiles = CDIV(N, 16) * (HDIM / 16);
  k_gemm_bias<<<CDIV(projTiles, 4), 128, 0, stream>>>(x_bf, Wq_bf, Wq_b, q_f, N, DDIM, HDIM, 1.0f);
  k_gemm_bias<<<CDIV(projTiles, 4), 128, 0, stream>>>(x_bf, Wk_bf, Wk_b, k_f, N, DDIM, HDIM, 1.0f);
  k_normalize<<<CDIV(N * NHEADS * 32, tb), tb, 0, stream>>>(q_f, k_f, qs_bf, kk_bf, N * NHEADS);

  k_zero_f32<<<CDIV(N, tb), tb, 0, stream>>>(deg, N);
  k_count_deg<<<CDIV(E, tb), tb, 0, stream>>>(erow, E, deg);
  k_edge_w<<<CDIV(E, tb), tb, 0, stream>>>(erow, ecol, deg, wE, E);
  k_zero_f32<<<CDIV(N * DDIM, tb), tb, 0, stream>>>(msg, N * DDIM);
  k_msg<<<(int)CDIV((long long)E * DDIM, tb), tb, 0, stream>>>(erow, eatt, bond, wE, msg, E);

  k_init<<<CDIV(N * HDIM, tb), tb, 0, stream>>>(x, cur, acc, cur_bf, N);

  // ---- K_ORDER propagation iterations ----
  for (int it = 0; it < KORDER; ++it) {
    float* gcn   = q_f;   // free after normalize
    float* numer = k_f;   // free after normalize
    k_zero_f32<<<CDIV(N * HDIM, tb), tb, 0, stream>>>(gcn, N * HDIM);
    k_gcn<<<(int)CDIV((long long)E * HDIM, tb), tb, 0, stream>>>(erow, ecol, wE, cur, gcn, E);
    k_sums<<<CDIV(B * HDIM, tb), tb, 0, stream>>>(ptrb, kk_bf, cur, ksum, vsum, B);
    k_kv<<<B * NHEADS, 128, 0, stream>>>(ptrb, kk_bf, cur_bf, kv, B);
    k_f32_to_bf16<<<CDIV(B * NHEADS * DDIM * DDIM, tb), tb, 0, stream>>>(
        kv, kv_bf, B * NHEADS * DDIM * DDIM);
    k_qkv<<<B * MAXNODE_TILES, 128, 0, stream>>>(ptrb, qs_bf, kv_bf, numer, MAXNODE_TILES);
    k_combine<<<N, HDIM, 0, stream>>>(qs_bf, ksum, vsum, numer, gcn, msg, batch, nn,
                                      cur, cur_bf, acc);
  }

  // ---- output projection (WMMA): out = (acc @ Wo^T + b) / NUM_HEADS ----
  k_f32_to_bf16<<<CDIV(N * HDIM, tb), tb, 0, stream>>>(acc, cur_bf, N * HDIM);
  int outTiles = CDIV(N, 16) * (DDIM / 16);
  k_gemm_bias<<<CDIV(outTiles, 4), 128, 0, stream>>>(cur_bf, Wo_bf, Wo_b, (float*)d_out,
                                                     N, HDIM, DDIM, 0.5f);
}
